// MultiHeadAttention_2714419331898
// MI455X (gfx1250) — matrix operand order repaired and checked by reading, not compiled, not run on hardware
//
#include <hip/hip_runtime.h>
#include <hip/hip_bf16.h>
#include <stdint.h>

// ---------------------------------------------------------------------------
// MultiHeadAttention (pre-LN, masked softmax, residual) for gfx1250 (MI455X)
// f16 WMMA (v_wmma_f32_16x16x32_f16, fp32 accum) for all 6 GEMMs, register
// blocking for operand reuse, TDM (tensor_load_to_lds) staging in softmax.
// HBM-bound on the mandatory 1 GiB fp32 attn output (~4 GB attn traffic =>
// ~180us floor @ 23.3 TB/s); f16 WMMA keeps compute far below that floor.
// Workspace requirement: 88 MiB.
// ---------------------------------------------------------------------------

typedef __attribute__((ext_vector_type(16))) _Float16 v16h;
typedef __attribute__((ext_vector_type(8)))  float    v8f;

#define HEADS   16
#define DM      1024
#define DK      64
#define BATCH   4
#define SEQ     2048
#define ROWS    (BATCH * SEQ)      /* 8192 */
#define NEGINF  (-1.0e9f)

union FragU { v16h v; _Float16 h[16]; uint32_t u[8]; };

// A-matrix 16x32 f16 K-striping (ISA 7.12.2): lane m=L&15, group g=L>>4,
// VGPR pair p holds K = (p>=4?16:0) + 8g + 2(p&3) + {0,1}
__device__ __forceinline__ int kpatA(int p, int g) {
  return ((p >= 4) ? 16 : 0) + g * 8 + 2 * (p & 3);
}
// B-matrix 32x16 f16: same K-striping as A, VGPR pair p holds
// K = (p>=4?16:0) + 8g + 2(p&3) + {0,1}
__device__ __forceinline__ int kpatB(int p, int g) {
  return ((p >= 4) ? 16 : 0) + g * 8 + 2 * (p & 3);
}

// A fragment: rows r0..r0+15, K-cols k0..k0+31, row-major f16, leading dim ld
__device__ __forceinline__ v16h fragA_f16(const _Float16* __restrict__ src, int ld, int r0, int k0) {
  int lane = threadIdx.x & 31;
  int m = lane & 15, g = lane >> 4;
  const _Float16* base = src + (size_t)(r0 + m) * ld + k0;
  FragU f;
#pragma unroll
  for (int p = 0; p < 8; ++p)
    f.u[p] = *(const uint32_t*)(base + kpatA(p, g));
  return f.v;
}

// A fragment gathered from fp32 source with on-the-fly f16 convert
__device__ __forceinline__ v16h fragA_f32(const float* __restrict__ src, int ld, int r0, int k0) {
  int lane = threadIdx.x & 31;
  int m = lane & 15, g = lane >> 4;
  const float* base = src + (size_t)(r0 + m) * ld + k0;
  FragU f;
#pragma unroll
  for (int p = 0; p < 8; ++p) {
    int kk = kpatA(p, g);
    f.h[2 * p]     = (_Float16)base[kk];
    f.h[2 * p + 1] = (_Float16)base[kk + 1];
  }
  return f.v;
}

// B fragment: operand stored N-major (srcT[n][k], contiguous in K)
__device__ __forceinline__ v16h fragB_f16(const _Float16* __restrict__ srcT, int ld, int n0, int k0) {
  int lane = threadIdx.x & 31;
  int n = lane & 15, g = lane >> 4;
  const _Float16* base = srcT + (size_t)(n0 + n) * ld + k0;
  FragU f;
#pragma unroll
  for (int p = 0; p < 8; ++p)
    f.u[p] = *(const uint32_t*)(base + kpatB(p, g));
  return f.v;
}

__device__ __forceinline__ v8f wmma_step(v16h a, v16h b, v8f c) {
  return __builtin_amdgcn_wmma_f32_16x16x32_f16(false, a, false, b, (short)0, c, false, false);
}

// ---------------------------------------------------------------------------
// K0: convert+transpose the 4 weight matrices to f16 W^T[n][k]
// ---------------------------------------------------------------------------
__global__ void wconv_kernel(const float* __restrict__ Wq, const float* __restrict__ Wk,
                             const float* __restrict__ Wv, const float* __restrict__ Wo,
                             _Float16* __restrict__ w16) {
  size_t t = (size_t)blockIdx.x * blockDim.x + threadIdx.x;  // 4 * 1M threads
  int w = (int)(t >> 20);
  int e = (int)(t & ((1u << 20) - 1));
  int k = e >> 10, n = e & 1023;
  const float* src = (w == 0) ? Wq : (w == 1) ? Wk : (w == 2) ? Wv : Wo;
  _Float16* dst = w16 + (size_t)w * DM * DM;
  dst[(size_t)n * DM + k] = (_Float16)src[(size_t)k * DM + n];
}

// ---------------------------------------------------------------------------
// K1: LayerNorm(q) row-wise -> f16   (one block per row, 256 thr, 4 elems/thr)
// ---------------------------------------------------------------------------
__global__ void ln_kernel(const float* __restrict__ q, const float* __restrict__ gamma,
                          const float* __restrict__ beta, _Float16* __restrict__ qn16) {
  __shared__ float red[256];
  int row = blockIdx.x;
  const float* x = q + (size_t)row * DM;
  int tid = threadIdx.x;
  float vals[4];
  float s = 0.f;
#pragma unroll
  for (int i = 0; i < 4; ++i) { vals[i] = x[tid + 256 * i]; s += vals[i]; }
  red[tid] = s; __syncthreads();
  for (int off = 128; off; off >>= 1) { if (tid < off) red[tid] += red[tid + off]; __syncthreads(); }
  float mu = red[0] * (1.f / DM);
  __syncthreads();
  float vs = 0.f;
#pragma unroll
  for (int i = 0; i < 4; ++i) { float d = vals[i] - mu; vs += d * d; }
  red[tid] = vs; __syncthreads();
  for (int off = 128; off; off >>= 1) { if (tid < off) red[tid] += red[tid + off]; __syncthreads(); }
  float inv = rsqrtf(red[0] * (1.f / DM) + 1e-6f);
#pragma unroll
  for (int i = 0; i < 4; ++i) {
    int c = tid + 256 * i;
    qn16[(size_t)row * DM + c] = (_Float16)(((vals[i] - mu) * inv) * gamma[c] + beta[c]);
  }
}

// ---------------------------------------------------------------------------
// K2: fused Q/K/V projections, 16(M)x64(N) per wave (A-frag reused 4x).
//   Q: qn16 @ Wq  -> qh16 [b][h][s][64], pre-scaled by 1/sqrt(64)
//   K: k_f32 @ Wk -> kh16 [b][h][s][64]
//   V: v_f32 @ Wv -> vh16t [b][h][64][s]   (K-major for the PV B-operand)
// ---------------------------------------------------------------------------
__global__ void proj_kernel(const _Float16* __restrict__ qn16, const float* __restrict__ k_in,
                            const float* __restrict__ v_in, const _Float16* __restrict__ w16,
                            _Float16* __restrict__ qh16, _Float16* __restrict__ kh16,
                            _Float16* __restrict__ vh16t) {
  int wave  = threadIdx.x >> 5;
  int which = blockIdx.x >> 10;                    // 0=Q 1=K 2=V (block-uniform)
  int tile  = ((blockIdx.x & 1023) << 3) + wave;   // 0..8191
  int tm = tile >> 4, tg = tile & 15;
  int r0 = tm * 16, n0 = tg * 64;
  const _Float16* Wt = w16 + (size_t)which * DM * DM;
  v8f acc[4] = {};
  for (int k0 = 0; k0 < DM; k0 += 32) {
    v16h a;
    if (which == 0)      a = fragA_f16(qn16, DM, r0, k0);
    else if (which == 1) a = fragA_f32(k_in, DM, r0, k0);
    else                 a = fragA_f32(v_in, DM, r0, k0);
#pragma unroll
    for (int j = 0; j < 4; ++j)
      acc[j] = wmma_step(a, fragB_f16(Wt, DM, n0 + j * 16, k0), acc[j]);
  }
  int lane = threadIdx.x & 31;
  int n = lane & 15, g = lane >> 4;
#pragma unroll
  for (int j = 0; j < 4; ++j) {
    int col = n0 + j * 16 + n, h = col >> 6, d = col & 63;
#pragma unroll
    for (int r = 0; r < 8; ++r) {
      int row = r0 + r + 8 * g;
      int b_ = row >> 11, s = row & 2047;
      if (which == 2) {
        vh16t[((size_t)(b_ * HEADS + h) * DK + d) * SEQ + s] = (_Float16)acc[j][r];
      } else {
        size_t idx = ((size_t)(b_ * HEADS + h) * SEQ + s) * DK + d;
        if (which == 0) qh16[idx] = (_Float16)(acc[j][r] * 0.125f);   // 1/sqrt(D_K)
        else            kh16[idx] = (_Float16)acc[j][r];
      }
    }
  }
}

// ---------------------------------------------------------------------------
// K3: logits = (Q/temp) @ K^T + key mask. 32x32 per wave (2x2 WMMA tiles).
// ---------------------------------------------------------------------------
__global__ void logits_kernel(const _Float16* __restrict__ qh16, const _Float16* __restrict__ kh16,
                              const int* __restrict__ mask, float* __restrict__ attn) {
  int wave = threadIdx.x >> 5;
  int tile = blockIdx.x * 8 + wave;                // 0..262143
  int bh = tile >> 12;
  int t  = tile & 4095;
  int tm = t >> 6, tn = t & 63;
  int r0 = tm * 32, c0 = tn * 32;
  const _Float16* qb = qh16 + (size_t)bh * SEQ * DK;
  const _Float16* kb = kh16 + (size_t)bh * SEQ * DK;   // rows of K == cols of K^T
  v8f acc[2][2] = {};
#pragma unroll
  for (int k0 = 0; k0 < DK; k0 += 32) {
    v16h a0 = fragA_f16(qb, DK, r0, k0);
    v16h a1 = fragA_f16(qb, DK, r0 + 16, k0);
    v16h b0 = fragB_f16(kb, DK, c0, k0);
    v16h b1 = fragB_f16(kb, DK, c0 + 16, k0);
    acc[0][0] = wmma_step(a0, b0, acc[0][0]);
    acc[0][1] = wmma_step(a0, b1, acc[0][1]);
    acc[1][0] = wmma_step(a1, b0, acc[1][0]);
    acc[1][1] = wmma_step(a1, b1, acc[1][1]);
  }
  int lane = threadIdx.x & 31;
  int n = lane & 15, g = lane >> 4;
  const int* mrow = mask + (bh >> 4) * SEQ;
  float* out = attn + (size_t)bh * SEQ * SEQ;
#pragma unroll
  for (int cj = 0; cj < 2; ++cj) {
    int col = c0 + cj * 16 + n;
    int mv = mrow[col];
#pragma unroll
    for (int ri = 0; ri < 2; ++ri) {
#pragma unroll
      for (int r = 0; r < 8; ++r) {
        int row = r0 + ri * 16 + r + 8 * g;
        out[(size_t)row * SEQ + col] = (mv == 0) ? NEGINF : acc[ri][cj][r];
      }
    }
  }
}

// ---------------------------------------------------------------------------
// K4: row-wise softmax in place. Row (8KB) is DMA'd into LDS by the Tensor
// Data Mover when available (tensor_load_to_lds + s_wait_tensorcnt).
// ---------------------------------------------------------------------------
__global__ void softmax_kernel(float* __restrict__ attn) {
  __shared__ float red[256];
  __shared__ __align__(16) float rowbuf[SEQ];      // 8 KB staging
  float* row = attn + (size_t)blockIdx.x * SEQ;
  int tid = threadIdx.x;

#if __has_builtin(__builtin_amdgcn_tensor_load_to_lds) && __has_builtin(__builtin_amdgcn_s_wait_tensorcnt)
  typedef uint32_t u32x4 __attribute__((ext_vector_type(4)));
  typedef int      i32x8 __attribute__((ext_vector_type(8)));
  typedef int      i32x4 __attribute__((ext_vector_type(4)));
  if (tid < 32) {                    // one wave issues the per-wave TDM op
    uint64_t ga = (uint64_t)(uintptr_t)row;
    uint32_t lo = (uint32_t)(uintptr_t)rowbuf;     // LDS aperture: low 32 bits
    u32x4 g0 = {};
    g0[0] = 1u;                                    // count=1 (valid descriptor)
    g0[1] = lo;                                    // lds_addr
    g0[2] = (uint32_t)ga;                          // global_addr[31:0]
    g0[3] = (uint32_t)(ga >> 32) | (2u << 30);     // global_addr[56:32] | type=2
    i32x8 g1 = {};
    g1[0] = 0x20000;                               // data_size=2 (4 bytes)
    g1[1] = (int)((SEQ & 0xFFFF) << 16);           // tensor_dim0[15:0]=2048
    g1[2] = (int)(1u << 16);                       // tensor_dim1[15:0]=1
    g1[3] = (int)((unsigned)SEQ << 16);            // tile_dim0=2048
    g1[4] = 1;                                     // tile_dim1=1, tile_dim2=0
    g1[5] = SEQ;                                   // tensor_dim0_stride lo
    i32x4 z4 = {};
    i32x8 z8 = {};
    // clang-23 / therock-10.0 form: (g0, g1, g2, g3, g4, cpol)
    __builtin_amdgcn_tensor_load_to_lds(g0, g1, z4, z4, z8, 0);
    __builtin_amdgcn_s_wait_tensorcnt((short)0);
  }
  __syncthreads();
  const float* src = rowbuf;
#else
  const float* src = row;
#endif

  float vals[8];
  float mx = -3.4e38f;
#pragma unroll
  for (int i = 0; i < 8; ++i) { vals[i] = src[tid + 256 * i]; mx = fmaxf(mx, vals[i]); }
  red[tid] = mx; __syncthreads();
  for (int off = 128; off; off >>= 1) { if (tid < off) red[tid] = fmaxf(red[tid], red[tid + off]); __syncthreads(); }
  mx = red[0]; __syncthreads();
  float s = 0.f;
#pragma unroll
  for (int i = 0; i < 8; ++i) { vals[i] = expf(vals[i] - mx); s += vals[i]; }
  red[tid] = s; __syncthreads();
  for (int off = 128; off; off >>= 1) { if (tid < off) red[tid] += red[tid + off]; __syncthreads(); }
  float inv = 1.0f / red[0];
#pragma unroll
  for (int i = 0; i < 8; ++i) row[tid + 256 * i] = vals[i] * inv;
}

// ---------------------------------------------------------------------------
// K5: out_h = attn @ V, 16(M)x64(N=full D_V) per wave: each fp32 attn
// A-fragment (the dominant HBM stream) is loaded ONCE and reused by 4 WMMAs.
// ---------------------------------------------------------------------------
__global__ void av_kernel(const float* __restrict__ attn, const _Float16* __restrict__ vh16t,
                          _Float16* __restrict__ oh16) {
  int wave = threadIdx.x >> 5;
  int tile = blockIdx.x * 8 + wave;                // 0..8191
  int bh = tile >> 7;
  int tm = tile & 127;
  int r0 = tm * 16;
  const float*    ab = attn  + (size_t)bh * SEQ * SEQ;
  const _Float16* vb = vh16t + (size_t)bh * DK * SEQ;
  v8f acc[4] = {};
  for (int k0 = 0; k0 < SEQ; k0 += 32) {
    v16h a = fragA_f32(ab, SEQ, r0, k0);
#pragma unroll
    for (int j = 0; j < 4; ++j)
      acc[j] = wmma_step(a, fragB_f16(vb, SEQ, j * 16, k0), acc[j]);
  }
  int lane = threadIdx.x & 31;
  int n = lane & 15, g = lane >> 4;
  int b_ = bh >> 4, h = bh & 15;
#pragma unroll
  for (int j = 0; j < 4; ++j) {
    int d = j * 16 + n;
#pragma unroll
    for (int r = 0; r < 8; ++r) {
      int s = r0 + r + 8 * g;
      oh16[(size_t)(b_ * SEQ + s) * DM + h * DK + d] = (_Float16)acc[j][r];
    }
  }
}

// ---------------------------------------------------------------------------
// K6: final = oh16 @ Wo + residual(q) -> fp32 out. 16x64 per wave.
// ---------------------------------------------------------------------------
__global__ void outproj_kernel(const _Float16* __restrict__ oh16, const _Float16* __restrict__ wo16t,
                               const float* __restrict__ q_in, float* __restrict__ out) {
  int wave = threadIdx.x >> 5;
  int tile = blockIdx.x * 8 + wave;                // 0..8191
  int tm = tile >> 4, tg = tile & 15;
  int r0 = tm * 16, n0 = tg * 64;
  v8f acc[4] = {};
  for (int k0 = 0; k0 < DM; k0 += 32) {
    v16h a = fragA_f16(oh16, DM, r0, k0);
#pragma unroll
    for (int j = 0; j < 4; ++j)
      acc[j] = wmma_step(a, fragB_f16(wo16t, DM, n0 + j * 16, k0), acc[j]);
  }
  int lane = threadIdx.x & 31;
  int n = lane & 15, g = lane >> 4;
#pragma unroll
  for (int j = 0; j < 4; ++j) {
    int col = n0 + j * 16 + n;
#pragma unroll
    for (int r = 0; r < 8; ++r) {
      int row = r0 + r + 8 * g;
      size_t idx = (size_t)row * DM + col;
      out[idx] = acc[j][r] + q_in[idx];
    }
  }
}

// ---------------------------------------------------------------------------
extern "C" void kernel_launch(void* const* d_in, const int* in_sizes, int n_in,
                              void* d_out, int out_size, void* d_ws, size_t ws_size,
                              hipStream_t stream) {
  const float* q     = (const float*)d_in[0];
  const float* k     = (const float*)d_in[1];
  const float* v     = (const float*)d_in[2];
  const int*   mask  = (const int*)d_in[3];
  const float* Wq    = (const float*)d_in[4];
  const float* Wk    = (const float*)d_in[5];
  const float* Wv    = (const float*)d_in[6];
  const float* Wo    = (const float*)d_in[7];
  const float* gamma = (const float*)d_in[8];
  const float* beta  = (const float*)d_in[9];

  float* out  = (float*)d_out;                     // [8192][1024] fp32
  float* attn = out + (size_t)ROWS * DM;           // [64][2048][2048] fp32

  // Workspace layout (88 MiB total)
  char* ws = (char*)d_ws;
  const size_t MB = 1ull << 20;
  _Float16* w16   = (_Float16*)(ws);               // 8 MiB: Wq^T,Wk^T,Wv^T,Wo^T f16
  _Float16* qn16  = (_Float16*)(ws + 8 * MB);      // 16 MiB: LayerNorm(q) f16
  _Float16* qh16  = (_Float16*)(ws + 24 * MB);     // 16 MiB: Q heads (pre-scaled)
  _Float16* kh16  = (_Float16*)(ws + 40 * MB);     // 16 MiB: K heads
  _Float16* vh16t = (_Float16*)(ws + 56 * MB);     // 16 MiB: V heads, K-major
  _Float16* oh16  = (_Float16*)(ws + 72 * MB);     // 16 MiB: attn@V, interleaved

  wconv_kernel  <<<16384,               256, 0, stream>>>(Wq, Wk, Wv, Wo, w16);
  ln_kernel     <<<ROWS,                256, 0, stream>>>(q, gamma, beta, qn16);
  proj_kernel   <<<3 * 1024,            256, 0, stream>>>(qn16, k, v, w16, qh16, kh16, vh16t);
  logits_kernel <<<32768,               256, 0, stream>>>(qh16, kh16, mask, attn);
  softmax_kernel<<<BATCH * HEADS * SEQ, 256, 0, stream>>>(attn);
  av_kernel     <<<1024,                256, 0, stream>>>(attn, vh16t, oh16);
  outproj_kernel<<<1024,                256, 0, stream>>>(oh16, w16 + 3ull * DM * DM, q, out);
}
